// TransformerUnit_79620103733527
// MI455X (gfx1250) — matrix operand order repaired and checked by reading, not compiled, not run on hardware
//
#include <hip/hip_runtime.h>
#include <hip/hip_bf16.h>

// ---------------------------------------------------------------------------
// Transformer-XL layer for MI455X (gfx1250).
//  - all matmuls via v_wmma_f32_16x16x32_bf16 (wave32 WMMA)
//  - attention fused: 16x2048 f32 scores + 16x2048 bf16 probs live in LDS
//    (~201KB of the 320KB WGP LDS), scores never touch HBM
//  - rel_shift resolved analytically: BD[i,j] = BDraw[i, j+L-1-i], and the
//    out-of-range raw columns coincide exactly with the causal mask
//  - weights pre-swizzled to [K/32][N][32] so every WMMA B-tile is a
//    contiguous 32B run per lane (2x global_load_b128)
// ---------------------------------------------------------------------------

#define L_  1024
#define B_  2
#define D_  1024
#define H_  16
#define DH_ 64
#define DFF_ 4096
#define MEM_ 1024
#define T_  2048
#define SCALE_ 0.125f   // 1/sqrt(64)

typedef __bf16 bf16;
typedef __attribute__((ext_vector_type(16))) __bf16 v16bf;
typedef __attribute__((ext_vector_type(8)))  float  v8f;

union V16 { v16bf v; uint4 q[2]; };

// ---- bf16 helpers (RNE f32->bf16) -----------------------------------------
__device__ __forceinline__ bf16 f2bf(float f) {
    union { float f; unsigned u; } x; x.f = f;
    unsigned u = x.u;
    u += 0x7FFFu + ((u >> 16) & 1u);
    union { unsigned short s; bf16 b; } y;
    y.s = (unsigned short)(u >> 16);
    return y.b;
}

// ---- WMMA wrapper ----------------------------------------------------------
__device__ __forceinline__ v8f wmma_bf16(v16bf a, v16bf b, v8f c) {
    return __builtin_amdgcn_wmma_f32_16x16x32_bf16(
        /*neg_a=*/false, a, /*neg_b=*/false, b,
        /*c_mod=*/(short)0, c, /*reuse_a=*/false, /*reuse_b=*/false);
}

// ---- operand loaders (contiguous-k layouts (k = 16*(l>>4)+s), wave32) --------------------------
// A 16x32 (MxK), row-major source, leading dim ld (elements):
//   lane l: row m = l&15; elems[0:16) = K[kh..kh+16),
//   kh = 16*(l>>4).  Two contiguous 16B chunks per lane.
__device__ __forceinline__ v16bf load_a(const bf16* __restrict__ p, int ld) {
    int lane = threadIdx.x & 31;
    int m = lane & 15;
    int kh = (lane >> 4) * 16;
    const bf16* r = p + (size_t)m * ld + kh;
    V16 u;
    u.q[0] = *(const uint4*)(r);
    u.q[1] = *(const uint4*)(r + 8);
    return u.v;
}

// B 32x16 (KxN): lane l holds column n = l&15, contiguous K-run
//   K = kb0..kb0+15 with kb0 = 16*(l>>4)   (same contiguous order as load_a)
// Transposed source (K contiguous): B[k,n] = p[n*ld + k] -> 32B contiguous.
__device__ __forceinline__ v16bf load_bt(const bf16* __restrict__ p, int ld) {
    int lane = threadIdx.x & 31;
    int n = lane & 15;
    int kb0 = (lane >> 4) * 16;
    const bf16* r = p + (size_t)n * ld + kb0;
    V16 u;
    u.q[0] = *(const uint4*)(r);
    u.q[1] = *(const uint4*)(r + 8);
    return u.v;
}

// guarded transposed load: column n0+lane-col, zeros when >= nlim (masked)
__device__ __forceinline__ v16bf load_bt_g(const bf16* __restrict__ p, int ld,
                                           int n0, int nlim, int d0) {
    int lane = threadIdx.x & 31;
    int n = n0 + (lane & 15);
    int kb0 = (lane >> 4) * 16;
    V16 u;
    if (n < nlim) {
        const bf16* r = p + (size_t)n * ld + d0 + kb0;
        u.q[0] = *(const uint4*)(r);
        u.q[1] = *(const uint4*)(r + 8);
    } else {
        u.q[0] = uint4{0, 0, 0, 0};
        u.q[1] = uint4{0, 0, 0, 0};
    }
    return u.v;
}

// packed-B source: swizzled [K/32][N][32]; tile (k0,n0) -> 32B contiguous
__device__ __forceinline__ v16bf load_bp(const bf16* __restrict__ pbase,
                                         int N, int k0, int n0) {
    int lane = threadIdx.x & 31;
    int n = n0 + (lane & 15);
    int kb0 = (lane >> 4) * 16;
    const bf16* r = pbase + ((size_t)(k0 >> 5) * N + n) * 32 + kb0;
    V16 u;
    u.q[0] = *(const uint4*)(r);
    u.q[1] = *(const uint4*)(r + 8);
    return u.v;
}

// ---------------------------------------------------------------------------
// Elementwise / packing kernels
// ---------------------------------------------------------------------------
__global__ void conv_f32_bf16(const float* __restrict__ src,
                              bf16* __restrict__ dst, size_t n) {
    size_t i = (size_t)blockIdx.x * blockDim.x + threadIdx.x;
    size_t stride = (size_t)gridDim.x * blockDim.x;
    for (; i < n; i += stride) dst[i] = f2bf(src[i]);
}

// convert + swizzle row-major f32 [K,N] into packed bf16 [K/32][N][32]
__global__ void conv_pack_bswz(const float* __restrict__ src,
                               bf16* __restrict__ dst, int K, int N) {
    size_t total = (size_t)K * N;
    size_t i = (size_t)blockIdx.x * blockDim.x + threadIdx.x;
    size_t stride = (size_t)gridDim.x * blockDim.x;
    for (; i < total; i += stride) {
        int k = (int)(i / N);
        int n = (int)(i - (size_t)k * N);
        dst[((size_t)(k >> 5) * N + n) * 32 + (k & 31)] = f2bf(src[i]);
    }
}

__global__ void zero_f32(float* __restrict__ p, size_t n) {
    size_t i = (size_t)blockIdx.x * blockDim.x + threadIdx.x;
    size_t stride = (size_t)gridDim.x * blockDim.x;
    for (; i < n; i += stride) p[i] = 0.0f;
}

// cb = bf16(concat(memory, x)) flattened [T*B, D]
__global__ void build_cb(const float* __restrict__ memory,
                         const float* __restrict__ x,
                         bf16* __restrict__ cb, size_t nmem, size_t ntot) {
    size_t i = (size_t)blockIdx.x * blockDim.x + threadIdx.x;
    size_t stride = (size_t)gridDim.x * blockDim.x;
    for (; i < ntot; i += stride) {
        float v = (i < nmem) ? memory[i] : x[i - nmem];
        cb[i] = f2bf(v);
    }
}

// split kv f32 [T*B, 2*H*DH] / r f32 [T*B, H*DH] into per-(b,h) bf16 buffers.
//   kb, rb: [B][H][T][DH] row-major (K^T / R^T tiles contiguous per lane)
//   vb    : [B][H][T/32][DH][32] packed (PV B-tiles contiguous per lane)
__global__ void pack_kvr(const float* __restrict__ kvf,
                         const float* __restrict__ rf,
                         bf16* __restrict__ kb, bf16* __restrict__ vb,
                         bf16* __restrict__ rb) {
    size_t n = (size_t)B_ * H_ * T_ * DH_;
    size_t i = (size_t)blockIdx.x * blockDim.x + threadIdx.x;
    size_t stride = (size_t)gridDim.x * blockDim.x;
    for (; i < n; i += stride) {
        int d = (int)(i & 63);
        int t = (int)((i >> 6) & 2047);
        int h = (int)((i >> 17) & 15);
        int b = (int)(i >> 21);
        size_t row = (size_t)t * B_ + b;
        size_t bh = (size_t)(b * H_ + h) * T_ * DH_;
        kb[i] = f2bf(kvf[row * (2 * H_ * DH_) + h * DH_ + d]);
        rb[i] = f2bf(rf[row * (H_ * DH_) + h * DH_ + d]);
        vb[bh + ((size_t)(t >> 5) * DH_ + d) * 32 + (t & 31)] =
            f2bf(kvf[row * (2 * H_ * DH_) + H_ * DH_ + h * DH_ + d]);
    }
}

// ---------------------------------------------------------------------------
// GEMM: C(f32)[M,N] = A(bf16)[M,K] @ Bswz(bf16); one wave per 16x64 tile
// ---------------------------------------------------------------------------
__global__ __launch_bounds__(32) void gemm_bf16_f32(
    const bf16* __restrict__ A, const bf16* __restrict__ Bm,
    float* __restrict__ C, int M, int N, int K) {
    int n0 = blockIdx.x * 64;
    int m0 = blockIdx.y * 16;
    v8f a0 = {}, a1 = {}, a2 = {}, a3 = {};
    for (int k0 = 0; k0 < K; k0 += 32) {
        __builtin_prefetch(A + (size_t)m0 * K + k0 + 64, 0, 1);
        v16bf a = load_a(A + (size_t)m0 * K + k0, K);
        a0 = wmma_bf16(a, load_bp(Bm, N, k0, n0), a0);
        a1 = wmma_bf16(a, load_bp(Bm, N, k0, n0 + 16), a1);
        a2 = wmma_bf16(a, load_bp(Bm, N, k0, n0 + 32), a2);
        a3 = wmma_bf16(a, load_bp(Bm, N, k0, n0 + 48), a3);
    }
    int lane = threadIdx.x & 31;
    int r0 = 8 * (lane >> 4), cc = lane & 15;
#pragma unroll
    for (int r = 0; r < 8; ++r) {
        size_t row = (size_t)(m0 + r0 + r);
        C[row * N + n0 + cc]      = a0[r];
        C[row * N + n0 + 16 + cc] = a1[r];
        C[row * N + n0 + 32 + cc] = a2[r];
        C[row * N + n0 + 48 + cc] = a3[r];
    }
}

// C(bf16) = relu(A@B + bias)  (FFN layer 1; output feeds next WMMA GEMM as A)
__global__ __launch_bounds__(32) void gemm_bias_relu_bf16(
    const bf16* __restrict__ A, const bf16* __restrict__ Bm,
    const float* __restrict__ bias, bf16* __restrict__ C,
    int M, int N, int K) {
    int n0 = blockIdx.x * 64;
    int m0 = blockIdx.y * 16;
    v8f a0 = {}, a1 = {}, a2 = {}, a3 = {};
    for (int k0 = 0; k0 < K; k0 += 32) {
        __builtin_prefetch(A + (size_t)m0 * K + k0 + 64, 0, 1);
        v16bf a = load_a(A + (size_t)m0 * K + k0, K);
        a0 = wmma_bf16(a, load_bp(Bm, N, k0, n0), a0);
        a1 = wmma_bf16(a, load_bp(Bm, N, k0, n0 + 16), a1);
        a2 = wmma_bf16(a, load_bp(Bm, N, k0, n0 + 32), a2);
        a3 = wmma_bf16(a, load_bp(Bm, N, k0, n0 + 48), a3);
    }
    int lane = threadIdx.x & 31;
    int r0 = 8 * (lane >> 4), cc = lane & 15;
#pragma unroll
    for (int r = 0; r < 8; ++r) {
        size_t row = (size_t)(m0 + r0 + r);
        float v0 = fmaxf(a0[r] + bias[n0 + cc], 0.0f);
        float v1 = fmaxf(a1[r] + bias[n0 + 16 + cc], 0.0f);
        float v2 = fmaxf(a2[r] + bias[n0 + 32 + cc], 0.0f);
        float v3 = fmaxf(a3[r] + bias[n0 + 48 + cc], 0.0f);
        C[row * N + n0 + cc]      = f2bf(v0);
        C[row * N + n0 + 16 + cc] = f2bf(v1);
        C[row * N + n0 + 32 + cc] = f2bf(v2);
        C[row * N + n0 + 48 + cc] = f2bf(v3);
    }
}

// ---------------------------------------------------------------------------
// Fused rel-attention. Block = (16 query rows, one b, one h), 4 waves.
// LDS: f32 score[16][2048] (128KB) + bf16 prob[16][2048] (64KB) + scratch.
// ---------------------------------------------------------------------------
__global__ __launch_bounds__(128) void attn_kernel(
    const float* __restrict__ qf,      // [L*B, H*DH] f32
    const bf16* __restrict__ kb,       // [B][H][T][DH]
    const bf16* __restrict__ vb,       // [B][H][T/32][DH][32] packed
    const bf16* __restrict__ rb,       // [B][H][T][DH]
    const float* __restrict__ ubias,   // [H,DH]
    const float* __restrict__ vbias,   // [H,DH]
    float* __restrict__ vecbuf,        // [L*B, H*DH] f32
    float* __restrict__ attnm) {       // [L,T] f32 (atomic accum)
    extern __shared__ float smem[];
    float* sc  = smem;                     // 16*2048 f32
    float* bds = sc + 16 * T_;             // 4 * 16*32 f32
    float* red = bds + 4 * 16 * 32;        // 256 f32
    bf16*  pb  = (bf16*)(red + 256);       // 16*2048 bf16

    const int i0 = blockIdx.x * 16;
    const int h  = blockIdx.y;
    const int b  = blockIdx.z;
    const int wave = threadIdx.x >> 5;
    const int lane = threadIdx.x & 31;
    const int m  = lane & 15;
    const int kh = (lane >> 4) * 16;

    // ---- build q+u and q+v A-tiles (two K-halves of DH=64) ----
    const float* qrow = qf + ((size_t)(i0 + m) * B_ + b) * (H_ * DH_) + h * DH_;
    const float* up = ubias + h * DH_;
    const float* vp = vbias + h * DH_;
    v16bf qu[2], qv[2];
#pragma unroll
    for (int half = 0; half < 2; ++half) {
        int k0 = half * 32;
        V16 tu, tv;
#pragma unroll
        for (int e = 0; e < 8; ++e) {
            int kk  = k0 + kh + e;
            int kk2 = k0 + kh + 8 + e;
            float q1 = qrow[kk], q2 = qrow[kk2];
            tu.v[e]     = f2bf(q1 + up[kk]);
            tv.v[e]     = f2bf(q1 + vp[kk]);
            tu.v[e + 8] = f2bf(q2 + up[kk2]);
            tv.v[e + 8] = f2bf(q2 + vp[kk2]);
        }
        qu[half] = tu.v; qv[half] = tv.v;
    }

    const bf16* kp = kb + ((size_t)(b * H_ + h)) * T_ * DH_;
    const bf16* rp = rb + ((size_t)(b * H_ + h)) * T_ * DH_;

    // ---- scores: each wave handles every 4th 16-key tile ----
    const int r0 = 8 * (lane >> 4), cc = lane & 15;
    float* w = bds + wave * (16 * 32);
    for (int jt = wave; jt < T_ / 16; jt += 4) {
        int j0 = jt * 16;
        // AC = (q+u) @ K^T
        v8f ac = {};
        ac = wmma_bf16(qu[0], load_bt(kp + (size_t)j0 * DH_, DH_), ac);
        ac = wmma_bf16(qu[1], load_bt(kp + (size_t)j0 * DH_ + 32, DH_), ac);
        // BD pair over r columns [base, base+31]
        int base = j0 + (L_ - 16) - i0;
        v8f bd0 = {}, bd1 = {};
        bd0 = wmma_bf16(qv[0], load_bt_g(rp, DH_, base, T_, 0), bd0);
        bd0 = wmma_bf16(qv[1], load_bt_g(rp, DH_, base, T_, 32), bd0);
        bd1 = wmma_bf16(qv[0], load_bt_g(rp, DH_, base + 16, T_, 0), bd1);
        bd1 = wmma_bf16(qv[1], load_bt_g(rp, DH_, base + 16, T_, 32), bd1);
#pragma unroll
        for (int r = 0; r < 8; ++r) {
            w[(r0 + r) * 32 + cc]      = bd0[r];
            w[(r0 + r) * 32 + 16 + cc] = bd1[r];
        }
        // combine with per-row shifted diagonal, mask, scale
#pragma unroll
        for (int r = 0; r < 8; ++r) {
            int t = r0 + r, s = cc;
            float val = ac[r] + w[t * 32 + (s + 15 - t)];
            int j = j0 + s, i = i0 + t;
            val = (j > i + MEM_) ? -1e30f : val * SCALE_;
            sc[(size_t)t * T_ + j] = val;
        }
    }
    __syncthreads();

    // ---- softmax over T per row (8 threads x 256 cols per row) ----
    {
        int row = threadIdx.x >> 3;
        int part = threadIdx.x & 7;
        float* srow = sc + (size_t)row * T_;
        bf16* prow = pb + (size_t)row * T_;
        float mx = -1e30f;
        for (int j = part * 256; j < part * 256 + 256; ++j)
            mx = fmaxf(mx, srow[j]);
        red[row * 8 + part] = mx;
        __syncthreads();
        if (part == 0) {
            float m2 = red[row * 8];
            for (int p2 = 1; p2 < 8; ++p2) m2 = fmaxf(m2, red[row * 8 + p2]);
            red[row * 8] = m2;
        }
        __syncthreads();
        float rmax = red[row * 8];
        float ssum = 0.0f;
        for (int j = part * 256; j < part * 256 + 256; ++j) {
            float e = __expf(srow[j] - rmax);
            srow[j] = e;
            ssum += e;
        }
        __syncthreads();
        red[row * 8 + part] = ssum;
        __syncthreads();
        if (part == 0) {
            float s2 = 0.0f;
            for (int p2 = 0; p2 < 8; ++p2) s2 += red[row * 8 + p2];
            red[row * 8] = 1.0f / s2;
        }
        __syncthreads();
        float inv = red[row * 8];
        float* am = attnm + (size_t)(i0 + row) * T_;
        for (int j = part * 256; j < part * 256 + 256; ++j) {
            float p = srow[j] * inv;
            prow[j] = f2bf(p);
            atomicAdd(&am[j], p * (1.0f / (B_ * H_)));
        }
    }
    __syncthreads();

    // ---- vec = prob @ V; wave w owns 16 of the DH=64 output columns ----
    {
        int n0 = wave * 16;
        const bf16* vvp = vb + ((size_t)(b * H_ + h)) * T_ * DH_;
        v8f acc = {};
        for (int k0 = 0; k0 < T_; k0 += 32) {
            v16bf a = load_a(pb + (size_t)m * T_ + k0, T_);
            acc = wmma_bf16(a, load_bp(vvp, DH_, k0, n0), acc);
        }
#pragma unroll
        for (int r = 0; r < 8; ++r) {
            vecbuf[((size_t)(i0 + r0 + r) * B_ + b) * (H_ * DH_) +
                   h * DH_ + n0 + cc] = acc[r];
        }
    }
}

// ---------------------------------------------------------------------------
// LayerNorm over D=1024 per row: y = LN(A + Bv (+ bias)); optional bf16 copy
// ---------------------------------------------------------------------------
__global__ __launch_bounds__(256) void ln_kernel(
    const float* __restrict__ A, const float* __restrict__ Bv,
    const float* __restrict__ bias, const float* __restrict__ g,
    const float* __restrict__ be, float* __restrict__ outf,
    bf16* __restrict__ outb) {
    __shared__ float s1[256], s2[256];
    const size_t row = blockIdx.x;
    const int tid = threadIdx.x;
    float vals[4];
    float sum = 0.0f;
#pragma unroll
    for (int t = 0; t < 4; ++t) {
        int c = tid + t * 256;
        float v = A[row * D_ + c] + Bv[row * D_ + c] + (bias ? bias[c] : 0.0f);
        vals[t] = v;
        sum += v;
    }
    s1[tid] = sum;
    __syncthreads();
    for (int off = 128; off > 0; off >>= 1) {
        if (tid < off) s1[tid] += s1[tid + off];
        __syncthreads();
    }
    float mu = s1[0] * (1.0f / D_);
    float sq = 0.0f;
#pragma unroll
    for (int t = 0; t < 4; ++t) {
        float d = vals[t] - mu;
        sq += d * d;
    }
    s2[tid] = sq;
    __syncthreads();
    for (int off = 128; off > 0; off >>= 1) {
        if (tid < off) s2[tid] += s2[tid + off];
        __syncthreads();
    }
    float rstd = rsqrtf(s2[0] * (1.0f / D_) + 1e-5f);
#pragma unroll
    for (int t = 0; t < 4; ++t) {
        int c = tid + t * 256;
        float y = (vals[t] - mu) * rstd * g[c] + be[c];
        outf[row * D_ + c] = y;
        if (outb) outb[row * D_ + c] = f2bf(y);
    }
}

// ---------------------------------------------------------------------------
// Host orchestration
// ---------------------------------------------------------------------------
extern "C" void kernel_launch(void* const* d_in, const int* in_sizes, int n_in,
                              void* d_out, int out_size, void* d_ws,
                              size_t ws_size, hipStream_t stream) {
    (void)in_sizes; (void)n_in; (void)out_size; (void)ws_size;
    const float* x       = (const float*)d_in[0];   // [L,B,D]
    const float* pos_emb = (const float*)d_in[1];   // [T,B,D]
    const float* pbu     = (const float*)d_in[2];   // [H,DH]
    const float* pbv     = (const float*)d_in[3];   // [H,DH]
    const float* memory  = (const float*)d_in[4];   // [MEM,B,D]
    const float* Wq      = (const float*)d_in[5];   // [D,H*DH]
    const float* Wkv     = (const float*)d_in[6];   // [D,2*H*DH]
    const float* Wo      = (const float*)d_in[7];   // [H*DH,D]
    const float* Wrel    = (const float*)d_in[8];   // [D,H*DH]
    const float* ln1_g   = (const float*)d_in[9];
    const float* ln1_b   = (const float*)d_in[10];
    const float* W1      = (const float*)d_in[11];  // [D,DFF]
    const float* b1      = (const float*)d_in[12];
    const float* W2      = (const float*)d_in[13];  // [DFF,D]
    const float* b2      = (const float*)d_in[14];
    const float* ln2_g   = (const float*)d_in[15];
    const float* ln2_b   = (const float*)d_in[16];
    // d_in[17] = mask: recomputed analytically on device.

    float* outp  = (float*)d_out;                 // [L,B,D]
    float* attnm = outp + (size_t)L_ * B_ * D_;   // [L,T]

    // ---- bump allocator over workspace ----
    char* wsp = (char*)d_ws;
    auto alloc = [&](size_t bytes) -> void* {
        void* p = (void*)wsp;
        wsp += (bytes + 255) & ~(size_t)255;
        return p;
    };
    const size_t LB = (size_t)L_ * B_;   // 2048 rows
    const size_t TB = (size_t)T_ * B_;   // 4096 rows

    bf16* xb    = (bf16*)alloc(LB * D_ * 2);
    bf16* cb    = (bf16*)alloc(TB * D_ * 2);
    bf16* peb   = (bf16*)alloc(TB * D_ * 2);
    bf16* Wqb   = (bf16*)alloc((size_t)D_ * H_ * DH_ * 2);        // swizzled
    bf16* Wkvb  = (bf16*)alloc((size_t)D_ * 2 * H_ * DH_ * 2);    // swizzled
    bf16* Wrelb = (bf16*)alloc((size_t)D_ * H_ * DH_ * 2);        // swizzled
    bf16* Wob   = (bf16*)alloc((size_t)H_ * DH_ * D_ * 2);        // swizzled
    bf16* W1b   = (bf16*)alloc((size_t)D_ * DFF_ * 2);            // swizzled
    bf16* W2b   = (bf16*)alloc((size_t)DFF_ * D_ * 2);            // swizzled
    float* qf   = (float*)alloc(LB * H_ * DH_ * 4);
    float* kvf  = (float*)alloc(TB * 2 * H_ * DH_ * 4);
    float* rf   = (float*)alloc(TB * H_ * DH_ * 4);
    bf16* kb    = (bf16*)alloc((size_t)B_ * H_ * T_ * DH_ * 2);
    bf16* vbuf  = (bf16*)alloc((size_t)B_ * H_ * T_ * DH_ * 2);   // packed
    bf16* rbuf  = (bf16*)alloc((size_t)B_ * H_ * T_ * DH_ * 2);
    float* vecf = (float*)alloc(LB * H_ * DH_ * 4);
    bf16* vecb  = (bf16*)alloc(LB * H_ * DH_ * 2);
    float* aof  = (float*)alloc(LB * D_ * 4);
    float* o1f  = (float*)alloc(LB * D_ * 4);
    bf16* o1b   = (bf16*)alloc(LB * D_ * 2);
    bf16* h1b   = (bf16*)alloc(LB * DFF_ * 2);
    float* fff  = (float*)alloc(LB * D_ * 4);

    const int CB = 256;
    auto cgrid = [](size_t n) { return (unsigned)((n + 255) / 256); };

    // 1) zero attn accumulator (d_out is poisoned by harness)
    zero_f32<<<cgrid((size_t)L_ * T_), CB, 0, stream>>>(attnm, (size_t)L_ * T_);

    // 2) fp32 -> bf16 conversions (activations plain, weights swizzled)
    conv_f32_bf16<<<cgrid(LB * D_), CB, 0, stream>>>(x, xb, LB * D_);
    conv_f32_bf16<<<cgrid(TB * D_), CB, 0, stream>>>(pos_emb, peb, TB * D_);
    conv_pack_bswz<<<cgrid((size_t)D_ * H_ * DH_), CB, 0, stream>>>(Wq, Wqb, D_, H_ * DH_);
    conv_pack_bswz<<<cgrid((size_t)D_ * 2 * H_ * DH_), CB, 0, stream>>>(Wkv, Wkvb, D_, 2 * H_ * DH_);
    conv_pack_bswz<<<cgrid((size_t)D_ * H_ * DH_), CB, 0, stream>>>(Wrel, Wrelb, D_, H_ * DH_);
    conv_pack_bswz<<<cgrid((size_t)H_ * DH_ * D_), CB, 0, stream>>>(Wo, Wob, H_ * DH_, D_);
    conv_pack_bswz<<<cgrid((size_t)D_ * DFF_), CB, 0, stream>>>(W1, W1b, D_, DFF_);
    conv_pack_bswz<<<cgrid((size_t)DFF_ * D_), CB, 0, stream>>>(W2, W2b, DFF_, D_);
    build_cb<<<cgrid(TB * D_), CB, 0, stream>>>(memory, x, cb,
                                                (size_t)MEM_ * B_ * D_, TB * D_);

    // 3) projections: q, kv, r  (all WMMA)
    gemm_bf16_f32<<<dim3((H_ * DH_) / 64, LB / 16), 32, 0, stream>>>(
        xb, Wqb, qf, (int)LB, H_ * DH_, D_);
    gemm_bf16_f32<<<dim3((2 * H_ * DH_) / 64, TB / 16), 32, 0, stream>>>(
        cb, Wkvb, kvf, (int)TB, 2 * H_ * DH_, D_);
    gemm_bf16_f32<<<dim3((H_ * DH_) / 64, TB / 16), 32, 0, stream>>>(
        peb, Wrelb, rf, (int)TB, H_ * DH_, D_);

    // 4) head-major bf16 packing for attention
    pack_kvr<<<cgrid((size_t)B_ * H_ * T_ * DH_), CB, 0, stream>>>(
        kvf, rf, kb, vbuf, rbuf);

    // 5) fused attention (scores + probs live entirely in LDS)
    size_t shbytes = (16 * (size_t)T_ + 4 * 16 * 32 + 256) * sizeof(float) +
                     (size_t)16 * T_ * sizeof(bf16);
    attn_kernel<<<dim3(L_ / 16, H_, B_), 128, shbytes, stream>>>(
        qf, kb, vbuf, rbuf, pbu, pbv, vecf, attnm);

    // 6) output projection + LN1
    conv_f32_bf16<<<cgrid(LB * H_ * DH_), CB, 0, stream>>>(vecf, vecb,
                                                           LB * H_ * DH_);
    gemm_bf16_f32<<<dim3(D_ / 64, LB / 16), 32, 0, stream>>>(
        vecb, Wob, aof, (int)LB, D_, H_ * DH_);
    ln_kernel<<<(unsigned)LB, 256, 0, stream>>>(x, aof, nullptr, ln1_g, ln1_b,
                                                o1f, o1b);

    // 7) FFN: relu(o1 @ W1 + b1) @ W2 + b2, then LN2 -> d_out
    gemm_bias_relu_bf16<<<dim3(DFF_ / 64, LB / 16), 32, 0, stream>>>(
        o1b, W1b, b1, h1b, (int)LB, DFF_, D_);
    gemm_bf16_f32<<<dim3(D_ / 64, LB / 16), 32, 0, stream>>>(
        h1b, W2b, fff, (int)LB, D_, DFF_);
    ln_kernel<<<(unsigned)LB, 256, 0, stream>>>(o1f, fff, b2, ln2_g, ln2_b,
                                                outp, nullptr);
}
